// MHA_18184891531644
// MI455X (gfx1250) — matrix operand order repaired and checked by reading, not compiled, not run on hardware
//
#include <hip/hip_runtime.h>
#include <hip/hip_bf16.h>

typedef __attribute__((ext_vector_type(16))) __bf16 v16bf;
typedef __attribute__((ext_vector_type(8)))  __bf16 v8bf;
typedef __attribute__((ext_vector_type(8)))  unsigned short v8us;
typedef __attribute__((ext_vector_type(8)))  float  v8f;
typedef __attribute__((ext_vector_type(4)))  int    v4i;
typedef __attribute__((ext_vector_type(4)))  unsigned int v4u;
typedef __attribute__((ext_vector_type(8)))  int    v8i;

#define B_  4
#define T_  2048
#define C_  2048
#define H_  16
#define KV_ 4
#define HD_ 128
#define G_  (H_ / KV_)

#define AS1 __attribute__((address_space(1)))
#define AS3 __attribute__((address_space(3)))

// ---------------------------------------------------------------------------
// CDNA5 feature probes (file compiles either way)
// ---------------------------------------------------------------------------
#if defined(__has_builtin)
#if __has_builtin(__builtin_amdgcn_global_load_async_to_lds_b128)
#define HAVE_ASYNC_LDS 1
#endif
#if __has_builtin(__builtin_amdgcn_tensor_load_to_lds) && \
    __has_builtin(__builtin_amdgcn_s_wait_tensorcnt)
#define HAVE_TDM 1
#endif
#endif

__device__ __forceinline__ void copy16_g2l(const __bf16* g, __bf16* l) {
#if defined(HAVE_ASYNC_LDS)
    __builtin_amdgcn_global_load_async_to_lds_b128(
        (AS1 v4i*)(unsigned long long)g,
        (AS3 v4i*)(unsigned long long)l, 0, 0);
#else
    *(v8bf*)l = *(const v8bf*)g;
#endif
}

__device__ __forceinline__ void async_fence() {
#if defined(HAVE_ASYNC_LDS)
#if defined(__has_builtin) && __has_builtin(__builtin_amdgcn_s_wait_asynccnt)
    __builtin_amdgcn_s_wait_asynccnt(0);
#else
    asm volatile("s_wait_asynccnt 0" ::: "memory");
#endif
#endif
}

// butterfly-xor reduction step within 16-lane groups: one ds_swizzle_b32
template <int MASK>
__device__ __forceinline__ float swzx(float v) {
    return __int_as_float(__builtin_amdgcn_ds_swizzle(
        __float_as_int(v), (MASK << 10) | 0x1F));
}

// ---------------------------------------------------------------------------
// WMMA helpers (CDNA5 gfx1250, wave32)
// ---------------------------------------------------------------------------
__device__ __forceinline__ v8f wmma_bf16(v16bf a, v16bf b, v8f c) {
    return __builtin_amdgcn_wmma_f32_16x16x32_bf16(
        false, a, false, b, (short)0, c, false, false);
}

__device__ __forceinline__ v16bf make_frag(v8bf lo, v8bf hi) {
    v16bf r;
#pragma unroll
    for (int i = 0; i < 8; ++i) { r[i] = lo[i]; r[i + 8] = hi[i]; }
    return r;
}

// A-matrix 16x32 bf16 fragment: lane L holds row (L&15);
// elems 0-7 = K {kk..kk+7}, elems 8-15 = K {16+kk..}, kk=(L>>4)*8.
__device__ __forceinline__ v16bf load_afrag(const __bf16* base, int rstride, int lane) {
    const int row = lane & 15;
    const int kk  = (lane >> 4) * 8;
    const __bf16* p = base + row * rstride;
    return make_frag(*(const v8bf*)(p + kk), *(const v8bf*)(p + 16 + kk));
}

// B-matrix 32x16 bf16 fragment: lane L holds column (L&15);
// elems 0-7 = K {kk..kk+7}, elems 8-15 = K {16+kk..}, kk=(L>>4)*8.
__device__ __forceinline__ v16bf load_bfrag(const __bf16* base, int rstride, int lane) {
    const int col = lane & 15;
    const int ko  = (lane >> 4) * 8;
    const __bf16* p = base + col * rstride + ko;
    return make_frag(*(const v8bf*)p, *(const v8bf*)(p + 16));
}

// ---------------------------------------------------------------------------
// f32 -> bf16 conversion (vectorized x4)
// ---------------------------------------------------------------------------
__global__ void cvt_f32_bf16(const float* __restrict__ in, __bf16* __restrict__ out, int n) {
    int i = (blockIdx.x * blockDim.x + threadIdx.x) * 4;
    if (i >= n) return;
    float4 v = *(const float4*)(in + i);
    out[i + 0] = (__bf16)v.x;
    out[i + 1] = (__bf16)v.y;
    out[i + 2] = (__bf16)v.z;
    out[i + 3] = (__bf16)v.w;
}

// ---------------------------------------------------------------------------
// RoPE in-place on (B*T, heads, HD) bf16 tensor
// ---------------------------------------------------------------------------
__global__ void rope_kernel(__bf16* __restrict__ t,
                            const float* __restrict__ cosp,
                            const float* __restrict__ sinp, int heads) {
    const int total = B_ * T_ * heads * (HD_ / 2);
    int idx = blockIdx.x * blockDim.x + threadIdx.x;
    if (idx >= total) return;
    const int i    = idx % (HD_ / 2);
    const int rest = idx / (HD_ / 2);
    const int hh   = rest % heads;
    const int bt   = rest / heads;
    const int tt   = bt % T_;
    __bf16* p = t + ((size_t)bt * heads + hh) * HD_ + 2 * i;
    const float tr = (float)p[0];
    const float ti = (float)p[1];
    const float c  = cosp[tt * (HD_ / 2) + i];
    const float s  = sinp[tt * (HD_ / 2) + i];
    p[0] = (__bf16)(tr * c - ti * s);
    p[1] = (__bf16)(tr * s + ti * c);
}

// ---------------------------------------------------------------------------
// GEMM: C[M,N] = A[M,K] * B[N,K]^T, bf16 in, f32 accumulate.
// Block: 256 thr (8 waves), tile 128(M) x 64(N), BK=64. Wave tile 32x32.
// ---------------------------------------------------------------------------
template <typename OT>
__global__ void gemm_nt(const __bf16* __restrict__ A, const __bf16* __restrict__ Bw,
                        OT* __restrict__ Cm, int M, int N, int K) {
    __shared__ __bf16 As[128][72];   // padded: 36-bank stride, conflict-free
    __shared__ __bf16 Bs[64][72];

    const int tid  = threadIdx.x;
    const int lane = tid & 31;
    const int wave = tid >> 5;
    const int wm   = wave >> 1;
    const int wn   = wave & 1;
    const int m0   = blockIdx.x * 128;
    const int n0   = blockIdx.y * 64;

    v8f acc[2][2] = {};

    for (int k0 = 0; k0 < K; k0 += 64) {
        {   // stage A: 128 x 64, 32 elems (64B) per thread
            const int row = tid >> 1, ko = (tid & 1) * 32;
            const __bf16* g = A + (size_t)(m0 + row) * K + k0 + ko;
#pragma unroll
            for (int j = 0; j < 4; ++j) copy16_g2l(g + j * 8, &As[row][ko + j * 8]);
        }
        {   // stage B: 64 x 64, 16 elems (32B) per thread
            const int row = tid >> 2, ko = (tid & 3) * 16;
            const __bf16* g = Bw + (size_t)(n0 + row) * K + k0 + ko;
#pragma unroll
            for (int j = 0; j < 2; ++j) copy16_g2l(g + j * 8, &Bs[row][ko + j * 8]);
        }
        async_fence();
        __syncthreads();

#pragma unroll
        for (int ks = 0; ks < 64; ks += 32) {
            v16bf a0 = load_afrag(&As[wm * 32][ks],      72, lane);
            v16bf a1 = load_afrag(&As[wm * 32 + 16][ks], 72, lane);
            v16bf b0 = load_bfrag(&Bs[wn * 32][ks],      72, lane);
            v16bf b1 = load_bfrag(&Bs[wn * 32 + 16][ks], 72, lane);
            acc[0][0] = wmma_bf16(a0, b0, acc[0][0]);
            acc[0][1] = wmma_bf16(a0, b1, acc[0][1]);
            acc[1][0] = wmma_bf16(a1, b0, acc[1][0]);
            acc[1][1] = wmma_bf16(a1, b1, acc[1][1]);
        }
        __syncthreads();
    }

    const int hgrp = lane >> 4, col = lane & 15;
#pragma unroll
    for (int i = 0; i < 2; ++i)
#pragma unroll
        for (int j = 0; j < 2; ++j) {
            const int mb = m0 + wm * 32 + i * 16;
            const int nb = n0 + wn * 32 + j * 16 + col;
#pragma unroll
            for (int r = 0; r < 8; ++r)
                Cm[(size_t)(mb + r + 8 * hgrp) * N + nb] = (OT)acc[i][j][r];
        }
}

// ---------------------------------------------------------------------------
// Flash attention (causal, GQA). Block = 4 waves, 64 q-rows; wave = 16 rows.
// K tile staged via TDM (tensor_load_to_lds) when available; Q via async
// loads; V transposed through VGPRs.
// ---------------------------------------------------------------------------
__global__ void attn_kernel(const __bf16* __restrict__ Q, const __bf16* __restrict__ Kb,
                            const __bf16* __restrict__ Vb, __bf16* __restrict__ O) {
    __shared__ __bf16 Qs[64][136];           // Q tile (q-rows x HD), padded
    __shared__ __bf16 Ks[32][136];           // K tile (keys x HD), padded (=TDM pad)
    __shared__ unsigned short VtU[128][40];  // V tile transposed (HD x keys)
    __shared__ __bf16 Ps[4][16][40];         // per-wave P bounce

    const int tid  = threadIdx.x;
    const int lane = tid & 31;
    const int wave = tid >> 5;
    const int qt   = blockIdx.x;
    const int hh   = blockIdx.y;
    const int bb   = blockIdx.z;
    const int kvh  = hh / G_;
    const int qrow0 = qt * 64 + wave * 16;
    const int hgrp = lane >> 4, col = lane & 15;
    const float scale = 0.08838834764831845f;  // 1/sqrt(128)

    {   // stage Q tile once: 64 rows x 128, 64 elems per thread
        const int row = tid >> 1, ko = (tid & 1) * 64;
        const __bf16* qp = Q + ((size_t)(bb * T_ + qt * 64 + row) * H_ + hh) * HD_ + ko;
#pragma unroll
        for (int j = 0; j < 8; ++j) copy16_g2l(qp + j * 8, &Qs[row][ko + j * 8]);
    }

    v8f acc[8] = {};
    float m_i[8], l_i[8];
#pragma unroll
    for (int r = 0; r < 8; ++r) { m_i[r] = -3.0e38f; l_i[r] = 0.0f; }

    async_fence();
    __syncthreads();

    const int nkt = qt * 2 + 2;  // causal upper bound for this block
    for (int kt = 0; kt < nkt; ++kt) {
        const int key0 = kt * 32;
#if defined(HAVE_TDM)
        if (tid == 0) {
            // D# (ISA 8.3/8.4): 2D tile 128x32 bf16, row stride KV*HD elems,
            // LDS pad: every 64 DWORDs (one 256B row) insert 4 DWORDs -> 136-elem stride
            const unsigned long long ga = (unsigned long long)(Kb +
                ((size_t)(bb * T_ + key0) * KV_ + kvh) * HD_);
            const unsigned int lds = (unsigned int)(unsigned long long)&Ks[0][0];
            v4u g0;
            g0[0] = 1u;                                     // count=1 (valid D#)
            g0[1] = lds;                                    // lds_addr
            g0[2] = (unsigned int)ga;                       // global_addr[31:0]
            g0[3] = (unsigned int)((ga >> 32) & 0x01FFFFFFull) | (2u << 30); // [56:32]|type=2
            v8i g1;
            g1[0] = (1 << 16)                               // data_size = 2B
                  | (1 << 20)                               // pad_enable
                  | (5 << 22)                               // pad_interval = 64 DW
                  | (3 << 25);                              // pad_amount = 4 DW
            g1[1] = (HD_ & 0xFFFF) << 16;                   // tensor_dim0 = 128
            g1[2] = 0;                                      // dim0 hi / dim1 lo16 = 0
            g1[3] = (int)((1u << 20) >> 16) | (HD_ << 16);  // dim1 hi | tile_dim0 = 128
            g1[4] = 32;                                     // tile_dim1 = 32 keys
            g1[5] = KV_ * HD_;                              // tensor_dim0_stride = 512
            g1[6] = 0;
            g1[7] = 0;
            v4i g2 = {}, g3 = {};
            v8i g4 = {};
            __builtin_amdgcn_tensor_load_to_lds(g0, g1, g2, g3, g4, 0);
        }
#else
        {   // stage K tile: 32 keys x 128, 32 elems per thread
            const int key = tid >> 2, doff = (tid & 3) * 32;
            const __bf16* kp =
                Kb + ((size_t)(bb * T_ + key0 + key) * KV_ + kvh) * HD_ + doff;
#pragma unroll
            for (int j = 0; j < 4; ++j) copy16_g2l(kp + j * 8, &Ks[key][doff + j * 8]);
        }
#endif
        {   // stage V transposed: thread handles key pair (t>>3)*2, 16 d values
            const int kp2 = (tid >> 3) * 2, de = (tid & 7) * 16;
            const unsigned short* v0 = (const unsigned short*)(Vb +
                ((size_t)(bb * T_ + key0 + kp2) * KV_ + kvh) * HD_ + de);
            const unsigned short* v1 = (const unsigned short*)(Vb +
                ((size_t)(bb * T_ + key0 + kp2 + 1) * KV_ + kvh) * HD_ + de);
            v8us a0 = *(const v8us*)v0,       a1 = *(const v8us*)(v0 + 8);
            v8us b0 = *(const v8us*)v1,       b1 = *(const v8us*)(v1 + 8);
#pragma unroll
            for (int e = 0; e < 8; ++e) {
                *(unsigned int*)&VtU[de + e][kp2] =
                    (unsigned int)a0[e] | ((unsigned int)b0[e] << 16);
                *(unsigned int*)&VtU[de + 8 + e][kp2] =
                    (unsigned int)a1[e] | ((unsigned int)b1[e] << 16);
            }
        }
#if defined(HAVE_TDM)
        if (tid == 0) __builtin_amdgcn_s_wait_tensorcnt(0);
#endif
        async_fence();
        __syncthreads();

        if (key0 <= qrow0 + 15) {
            // S = Q K^T over HD: two 16x16 N-tiles (keys 0-15, 16-31)
            v8f s0 = {}, s1 = {};
#pragma unroll
            for (int c = 0; c < 4; ++c) {
                v16bf qf  = load_afrag(&Qs[wave * 16][c * 32], 136, lane);
                v16bf bk0 = load_bfrag(&Ks[0][c * 32],  136, lane);
                v16bf bk1 = load_bfrag(&Ks[16][c * 32], 136, lane);
                s0 = wmma_bf16(qf, bk0, s0);
                s1 = wmma_bf16(qf, bk1, s1);
            }

            // online softmax; rows live in 16-lane halves of the wave
#pragma unroll
            for (int r = 0; r < 8; ++r) {
                const int qr  = qrow0 + r + 8 * hgrp;
                float t0 = (key0 + col      <= qr) ? s0[r] * scale : -3.0e38f;
                float t1 = (key0 + 16 + col <= qr) ? s1[r] * scale : -3.0e38f;
                float mx = fmaxf(t0, t1);
                mx = fmaxf(mx, swzx<1>(mx));
                mx = fmaxf(mx, swzx<2>(mx));
                mx = fmaxf(mx, swzx<4>(mx));
                mx = fmaxf(mx, swzx<8>(mx));
                const float nm = fmaxf(m_i[r], mx);
                const float al = __expf(m_i[r] - nm);
                m_i[r] = nm;
                t0 = __expf(t0 - nm);
                t1 = __expf(t1 - nm);
                float rs = t0 + t1;
                rs += swzx<1>(rs);
                rs += swzx<2>(rs);
                rs += swzx<4>(rs);
                rs += swzx<8>(rs);
                l_i[r] = l_i[r] * al + rs;
#pragma unroll
                for (int d = 0; d < 8; ++d) acc[d][r] *= al;
                // P tile -> per-wave LDS (same-wave DS ops stay ordered)
                Ps[wave][r + 8 * hgrp][col]      = (__bf16)t0;
                Ps[wave][r + 8 * hgrp][16 + col] = (__bf16)t1;
            }

            v16bf pf = load_afrag(&Ps[wave][0][0], 40, lane);
#pragma unroll
            for (int d = 0; d < 8; ++d) {
                v16bf vf = load_bfrag((const __bf16*)&VtU[d * 16][0], 40, lane);
                acc[d] = wmma_bf16(pf, vf, acc[d]);
            }
        }
        __syncthreads();
    }

#pragma unroll
    for (int d = 0; d < 8; ++d)
#pragma unroll
        for (int r = 0; r < 8; ++r) {
            const int qr = qrow0 + r + 8 * hgrp;
            O[((size_t)(bb * T_ + qr) * H_ + hh) * HD_ + d * 16 + col] =
                (__bf16)(acc[d][r] / l_i[r]);
        }
}

// ---------------------------------------------------------------------------
// launch
// ---------------------------------------------------------------------------
extern "C" void kernel_launch(void* const* d_in, const int* in_sizes, int n_in,
                              void* d_out, int out_size, void* d_ws, size_t ws_size,
                              hipStream_t stream) {
    const float* x    = (const float*)d_in[0];
    const float* cosp = (const float*)d_in[1];
    const float* sinp = (const float*)d_in[2];
    const float* wq   = (const float*)d_in[3];
    const float* wk   = (const float*)d_in[4];
    const float* wv   = (const float*)d_in[5];
    const float* wo   = (const float*)d_in[6];
    float* out = (float*)d_out;

    char* ws = (char*)d_ws;
    size_t off = 0;
    auto alloc = [&](size_t elems) -> __bf16* {
        __bf16* p = (__bf16*)(ws + off);
        off += ((elems * 2 + 255) / 256) * 256;
        return p;
    };
    const size_t BT = (size_t)B_ * T_;  // 8192 rows
    __bf16* xb  = alloc(BT * C_);
    __bf16* wqb = alloc((size_t)H_ * HD_ * C_);
    __bf16* wkb = alloc((size_t)KV_ * HD_ * C_);
    __bf16* wvb = alloc((size_t)KV_ * HD_ * C_);
    __bf16* wob = alloc((size_t)C_ * H_ * HD_);
    __bf16* qb  = alloc(BT * H_ * HD_);
    __bf16* kb  = alloc(BT * KV_ * HD_);
    __bf16* vb  = alloc(BT * KV_ * HD_);
    __bf16* ob  = xb;  // reuse: x no longer needed after projections

    auto cvt = [&](const float* src, __bf16* dst, size_t n) {
        const int thr = 256;
        const int blk = (int)((n / 4 + thr - 1) / thr);
        cvt_f32_bf16<<<blk, thr, 0, stream>>>(src, dst, (int)n);
    };
    cvt(x,  xb,  BT * C_);
    cvt(wq, wqb, (size_t)H_ * HD_ * C_);
    cvt(wk, wkb, (size_t)KV_ * HD_ * C_);
    cvt(wv, wvb, (size_t)KV_ * HD_ * C_);
    cvt(wo, wob, (size_t)C_ * H_ * HD_);

    const int M = B_ * T_;  // 8192
    gemm_nt<__bf16><<<dim3(M / 128, (H_ * HD_) / 64), 256, 0, stream>>>(
        xb, wqb, qb, M, H_ * HD_, C_);
    gemm_nt<__bf16><<<dim3(M / 128, (KV_ * HD_) / 64), 256, 0, stream>>>(
        xb, wkb, kb, M, KV_ * HD_, C_);
    gemm_nt<__bf16><<<dim3(M / 128, (KV_ * HD_) / 64), 256, 0, stream>>>(
        xb, wvb, vb, M, KV_ * HD_, C_);

    {
        const int tq = B_ * T_ * H_ * (HD_ / 2);
        rope_kernel<<<(tq + 255) / 256, 256, 0, stream>>>(qb, cosp, sinp, H_);
        const int tk = B_ * T_ * KV_ * (HD_ / 2);
        rope_kernel<<<(tk + 255) / 256, 256, 0, stream>>>(kb, cosp, sinp, KV_);
    }

    attn_kernel<<<dim3(T_ / 64, H_, B_), 128, 0, stream>>>(qb, kb, vb, ob);

    gemm_nt<float><<<dim3(M / 128, C_ / 64), 256, 0, stream>>>(
        ob, wob, out, M, C_, H_ * HD_);
}
